// FluxBitBlock_14474039787744
// MI455X (gfx1250) — 8-bit matrix operand order repaired and checked by reading, compiled, not run on hardware
//
#include <hip/hip_runtime.h>
#include <cstddef>
#include <cstdint>


#define D_   2048
#define S_   4096
#define C_   512
#define I_   8192
#define T_   (S_ + C_)
#define DD_  ((size_t)D_ * (size_t)D_)

typedef __attribute__((ext_vector_type(8))) int v8i;
typedef __attribute__((address_space(3))) char lds_char_t;

__device__ __forceinline__ float sigmoidf_(float x) { return 1.0f / (1.0f + expf(-x)); }
__device__ __forceinline__ float siluf_(float x)    { return x / (1.0f + expf(-x)); }

__device__ __forceinline__ unsigned lds_off_u32(void* p) {
    return (unsigned)(uintptr_t)(lds_char_t*)p;
}

#define ASYNC_B128(ldsoff, gptr)                                         \
    asm volatile("global_load_async_to_lds_b128 %0, %1, off"             \
                 :: "v"(ldsoff), "v"(gptr) : "memory")

__global__ void absmean_part_kernel(const float* __restrict__ w, size_t n,
                                    float* __restrict__ part) {
    __shared__ float red[256];
    float s = 0.0f;
    for (size_t i = (size_t)blockIdx.x * 256 + threadIdx.x; i < n;
         i += (size_t)gridDim.x * 256)
        s += fabsf(w[i]);
    red[threadIdx.x] = s;
    __syncthreads();
    for (int k = 128; k > 0; k >>= 1) {
        if ((int)threadIdx.x < k) red[threadIdx.x] += red[threadIdx.x + k];
        __syncthreads();
    }
    if (threadIdx.x == 0) part[blockIdx.x] = red[0];
}

__global__ void absmean_final_kernel(const float* __restrict__ part, int np,
                                     float invn, float* __restrict__ scales) {
    __shared__ float red[256];
    float s = 0.0f;
    for (int i = threadIdx.x; i < np; i += 256) s += part[i];
    red[threadIdx.x] = s;
    __syncthreads();
    for (int k = 128; k > 0; k >>= 1) {
        if ((int)threadIdx.x < k) red[threadIdx.x] += red[threadIdx.x + k];
        __syncthreads();
    }
    if (threadIdx.x == 0) {
        float mean = red[0] * invn;
        float dq = fmaxf(mean, 1e-5f);
        scales[0] = 1.0f / dq;
        scales[1] = dq;
    }
}

__global__ void wquant_kernel(const float* __restrict__ w, size_t n,
                              const float* __restrict__ scales,
                              signed char* __restrict__ q) {
    float sc = scales[0];
    for (size_t i = (size_t)blockIdx.x * 256 + threadIdx.x; i < n;
         i += (size_t)gridDim.x * 256) {
        float v = rintf(w[i] * sc);
        v = fminf(fmaxf(v, -1.0f), 1.0f);
        q[i] = (signed char)v;
    }
}

__global__ void rms_quant_kernel(const float* __restrict__ x, int ldx, int K,
                                 signed char* __restrict__ q,
                                 float* __restrict__ rowScale, float eps) {
    const int row = blockIdx.x;
    const float* xr = x + (size_t)row * ldx;
    __shared__ float red[256];

    float ss = 0.0f;
    for (int k = threadIdx.x; k < K; k += 256) { float v = xr[k]; ss += v * v; }
    red[threadIdx.x] = ss;
    __syncthreads();
    for (int s = 128; s > 0; s >>= 1) {
        if ((int)threadIdx.x < s) red[threadIdx.x] += red[threadIdx.x + s];
        __syncthreads();
    }
    const float rms = rsqrtf(red[0] / (float)K + eps);
    __syncthreads();

    float mx = 0.0f;
    for (int k = threadIdx.x; k < K; k += 256) mx = fmaxf(mx, fabsf(xr[k] * rms));
    red[threadIdx.x] = mx;
    __syncthreads();
    for (int s = 128; s > 0; s >>= 1) {
        if ((int)threadIdx.x < s)
            red[threadIdx.x] = fmaxf(red[threadIdx.x], red[threadIdx.x + s]);
        __syncthreads();
    }
    const float amax  = fmaxf(red[0], 1e-5f);
    const float scale = 127.0f / amax;
    if (threadIdx.x == 0) rowScale[row] = amax / 127.0f;

    for (int k = threadIdx.x; k < K; k += 256) {
        float v = rintf(xr[k] * rms * scale);
        v = fminf(fmaxf(v, -128.0f), 127.0f);
        q[(size_t)row * K + k] = (signed char)v;
    }
}

__global__ void ln_mod_kernel(const float* __restrict__ x,
                              const float* __restrict__ scale,
                              const float* __restrict__ shift,
                              float* __restrict__ out) {
    const int row = blockIdx.x;
    const float* xr = x + (size_t)row * D_;
    __shared__ float r1[256];
    __shared__ float r2[256];
    float s1 = 0.0f, s2 = 0.0f;
    for (int k = threadIdx.x; k < D_; k += 256) {
        float v = xr[k]; s1 += v; s2 += v * v;
    }
    r1[threadIdx.x] = s1; r2[threadIdx.x] = s2;
    __syncthreads();
    for (int s = 128; s > 0; s >>= 1) {
        if ((int)threadIdx.x < s) {
            r1[threadIdx.x] += r1[threadIdx.x + s];
            r2[threadIdx.x] += r2[threadIdx.x + s];
        }
        __syncthreads();
    }
    const float mu  = r1[0] / (float)D_;
    const float var = r2[0] / (float)D_ - mu * mu;
    const float rs  = rsqrtf(var + 1e-6f);
    for (int k = threadIdx.x; k < D_; k += 256) {
        out[(size_t)row * D_ + k] =
            (xr[k] - mu) * rs * (1.0f + scale[k]) + shift[k];
    }
}

__global__ void silu_vec_kernel(const float* __restrict__ x, float* __restrict__ y, int n) {
    int i = blockIdx.x * 256 + threadIdx.x;
    if (i < n) y[i] = siluf_(x[i]);
}

__global__ void embed_kernel(const float* __restrict__ st,
                             const float* __restrict__ w,
                             const float* __restrict__ b,
                             float* __restrict__ e) {
    __shared__ float sh[D_];
    for (int k = threadIdx.x; k < D_; k += 256) sh[k] = st[k];
    __syncthreads();
    int j = blockIdx.x * 256 + threadIdx.x;
    if (j < 6 * D_) {
        const float* wr = w + (size_t)j * D_;
        float acc = b[j];
        for (int k = 0; k < D_; ++k) acc += sh[k] * wr[k];
        e[j] = acc;
    }
}

__global__ void hgrn_kernel(const float* __restrict__ iL,
                            const float* __restrict__ fL,
                            float* __restrict__ o) {
    const int d = blockIdx.x * 256 + threadIdx.x;
    if (d >= D_) return;
    float h = 0.0f;
    for (int t = 0; t < T_; ++t) {
        const size_t idx = (size_t)t * D_ + d;
        const float fv = sigmoidf_(fL[idx]);
        const float iv = siluf_(iL[idx]) * (1.0f - fv);
        h = fv * h + iv;
        o[idx] = h;
    }
}

__global__ void gated_kernel(const float* __restrict__ o,
                             const float* __restrict__ g,
                             const float* __restrict__ gnw,
                             float* __restrict__ v) {
    const int row = blockIdx.x;
    const float* orow = o + (size_t)row * D_;
    __shared__ float red[256];
    float ss = 0.0f;
    for (int k = threadIdx.x; k < D_; k += 256) { float x = orow[k]; ss += x * x; }
    red[threadIdx.x] = ss;
    __syncthreads();
    for (int s = 128; s > 0; s >>= 1) {
        if ((int)threadIdx.x < s) red[threadIdx.x] += red[threadIdx.x + s];
        __syncthreads();
    }
    const float rf = rsqrtf(red[0] / (float)D_ + 1e-6f);
    for (int k = threadIdx.x; k < D_; k += 256) {
        const size_t idx = (size_t)row * D_ + k;
        v[idx] = orow[k] * rf * gnw[k] * siluf_(g[idx]);
    }
}

__global__ void glu_kernel(float* __restrict__ y, int M) {
    const size_t total = (size_t)M * I_;
    for (size_t idx = (size_t)blockIdx.x * 256 + threadIdx.x; idx < total;
         idx += (size_t)gridDim.x * 256) {
        const size_t t = idx / I_;
        const size_t i = idx - t * I_;
        float* row = y + t * (size_t)(2 * I_);
        row[i] = siluf_(row[i]) * row[I_ + i];
    }
}

__global__ __launch_bounds__(256) void gemm_i8_kernel(
    const signed char* __restrict__ Aq, const float* __restrict__ aScale,
    const signed char* __restrict__ Wq, const float* __restrict__ wScale,
    float* __restrict__ out, const float* __restrict__ res,
    const float* __restrict__ gate, int M, int N, int K) {
    __shared__ __align__(16) signed char lsA[2][128][80];
    __shared__ __align__(16) signed char lsB[2][128][80];

    const int mBlk = blockIdx.y * 128;
    const int nBlk = blockIdx.x * 128;
    const int tid  = threadIdx.x;
    const int lane = tid & 31;
    const int wv   = tid >> 5;
    const int r    = lane & 15;
    const int h    = lane >> 4;

    const int rw = tid >> 2;
    const int cc = (tid & 3) << 4;
    const signed char* gA0 = Aq + (size_t)(mBlk + rw) * K + cc;
    const signed char* gA1 = Aq + (size_t)(mBlk + 64 + rw) * K + cc;
    const signed char* gB0 = Wq + (size_t)(nBlk + rw) * K + cc;
    const signed char* gB1 = Wq + (size_t)(nBlk + 64 + rw) * K + cc;
    unsigned lA0[2], lA1[2], lB0[2], lB1[2];
#pragma unroll
    for (int b = 0; b < 2; ++b) {
        lA0[b] = lds_off_u32(&lsA[b][rw][cc]);
        lA1[b] = lds_off_u32(&lsA[b][64 + rw][cc]);
        lB0[b] = lds_off_u32(&lsB[b][rw][cc]);
        lB1[b] = lds_off_u32(&lsB[b][64 + rw][cc]);
    }

    v8i acc[8];
#pragma unroll
    for (int j = 0; j < 8; ++j)
#pragma unroll
        for (int e = 0; e < 8; ++e) acc[j][e] = 0;

    auto compute_tile = [&](int buf) {
        v8i afrag;
#pragma unroll
        for (int v = 0; v < 8; ++v) {
            const int s2 = v & 3;
            const int kk = ((v >> 2) << 5) + (h << 4) + ((v & 3) << 2);
            afrag[v] = *(const int*)&lsA[buf][wv * 16 + r][kk];
        }
        v8i bf[8];
#pragma unroll
        for (int j = 0; j < 8; ++j)
#pragma unroll
            for (int v = 0; v < 8; ++v) {
                const int kk = ((v >> 2) << 5) + (h << 4) + ((v & 3) << 2);
                bf[j][v] = *(const int*)&lsB[buf][j * 16 + r][kk];
            }
#pragma unroll
        for (int j = 0; j < 8; ++j)
            acc[j] = __builtin_amdgcn_wmma_i32_16x16x64_iu8(
                true, afrag, true, bf[j], acc[j], false, false);
    };

    ASYNC_B128(lA0[0], gA0);
    ASYNC_B128(lA1[0], gA1);
    ASYNC_B128(lB0[0], gB0);
    ASYNC_B128(lB1[0], gB1);

    int buf = 0;
    for (int kt = 0; kt < K - 64; kt += 64) {
        ASYNC_B128(lA0[buf ^ 1], gA0 + kt + 64);
        ASYNC_B128(lA1[buf ^ 1], gA1 + kt + 64);
        ASYNC_B128(lB0[buf ^ 1], gB0 + kt + 64);
        ASYNC_B128(lB1[buf ^ 1], gB1 + kt + 64);
        __builtin_prefetch(gB0 + kt + 128, 0, 1);
        asm volatile("s_wait_asynccnt 4" ::: "memory");
        __syncthreads();
        compute_tile(buf);
        __syncthreads();
        buf ^= 1;
    }
    asm volatile("s_wait_asynccnt 0" ::: "memory");
    __syncthreads();
    compute_tile(buf);

    const float wsc = wScale[0];
#pragma unroll
    for (int j = 0; j < 8; ++j) {
        const int col = nBlk + j * 16 + r;
#pragma unroll
        for (int g = 0; g < 8; ++g) {
            const int m = mBlk + wv * 16 + g + (h << 3);
            const float val = (float)acc[j][g] * aScale[m] * wsc;
            const size_t idx = (size_t)m * N + col;
            out[idx] = res ? (res[idx] + gate[col] * val) : val;
        }
    }
}

static void quant_weight(const float* w, size_t n, signed char* q,
                         float* scales, float* part, hipStream_t st) {
    absmean_part_kernel<<<1024, 256, 0, st>>>(w, n, part);
    absmean_final_kernel<<<1, 256, 0, st>>>(part, 1024, (float)(1.0 / (double)n), scales);
    wquant_kernel<<<2048, 256, 0, st>>>(w, n, scales, q);
}

static void gemm_i8(const signed char* Aq, const float* aS,
                    const signed char* Wq, const float* wS, float* out,
                    const float* res, const float* gate,
                    int M, int N, int K, hipStream_t st) {
    dim3 grid(N / 128, M / 128);
    gemm_i8_kernel<<<grid, 256, 0, st>>>(Aq, aS, Wq, wS, out, res, gate, M, N, K);
}

extern "C" void kernel_launch(void* const* d_in, const int* in_sizes, int n_in,
                              void* d_out, int out_size, void* d_ws, size_t ws_size,
                              hipStream_t stream) {
    (void)in_sizes; (void)n_in; (void)out_size; (void)ws_size;

    const float* hidden   = (const float*)d_in[0];
    const float* enc      = (const float*)d_in[1];
    const float* temb     = (const float*)d_in[2];
    const float* w_emb_s  = (const float*)d_in[3];
    const float* b_emb_s  = (const float*)d_in[4];
    const float* w_emb_c  = (const float*)d_in[5];
    const float* b_emb_c  = (const float*)d_in[6];
    const float* w_qkv_s  = (const float*)d_in[7];
    const float* w_out_s  = (const float*)d_in[8];
    const float* w_qkv_c  = (const float*)d_in[9];
    const float* w_out_c  = (const float*)d_in[10];
    const float* w_ifg    = (const float*)d_in[11];
    const float* w_o      = (const float*)d_in[12];
    const float* gnw      = (const float*)d_in[13];
    const float* w_gate_s = (const float*)d_in[14];
    const float* w_down_s = (const float*)d_in[15];
    const float* w_gate_c = (const float*)d_in[16];
    const float* w_down_c = (const float*)d_in[17];
    float* out = (float*)d_out;

    size_t cur = 0;
    auto alloc = [&](size_t bytes) -> void* {
        void* p = (char*)d_ws + cur;
        cur += (bytes + 255) & ~(size_t)255;
        return p;
    };
    const size_t DDb = DD_;
    const size_t GDb = (size_t)2 * I_ * D_;
    const size_t DIb = (size_t)D_ * I_;

    signed char* qQS = (signed char*)alloc(DDb);
    signed char* qQC = (signed char*)alloc(DDb);
    signed char* qI  = (signed char*)alloc(DDb);
    signed char* qF  = (signed char*)alloc(DDb);
    signed char* qG  = (signed char*)alloc(DDb);
    signed char* qO  = (signed char*)alloc(DDb);
    signed char* qOS = (signed char*)alloc(DDb);
    signed char* qOC = (signed char*)alloc(DDb);
    signed char* qGS = (signed char*)alloc(GDb);
    signed char* qDS = (signed char*)alloc(DIb);
    signed char* qGC = (signed char*)alloc(GDb);
    signed char* qDC = (signed char*)alloc(DIb);

    float* wsc  = (float*)alloc(24 * sizeof(float));
    float* part = (float*)alloc(1024 * sizeof(float));
    float* st_  = (float*)alloc(D_ * sizeof(float));
    float* e_s  = (float*)alloc(6 * D_ * sizeof(float));
    float* e_c  = (float*)alloc(6 * D_ * sizeof(float));
    float* nh   = (float*)alloc((size_t)S_ * D_ * sizeof(float));
    float* nc   = (float*)alloc((size_t)C_ * D_ * sizeof(float));
    float* qj   = (float*)alloc((size_t)T_ * D_ * sizeof(float));
    float* iL   = (float*)alloc((size_t)T_ * D_ * sizeof(float));
    float* fL   = (float*)alloc((size_t)T_ * D_ * sizeof(float));
    float* gL   = (float*)alloc((size_t)T_ * D_ * sizeof(float));
    float* oB   = (float*)alloc((size_t)T_ * D_ * sizeof(float));
    signed char* aq = (signed char*)alloc((size_t)S_ * I_);
    float* aSc  = (float*)alloc(T_ * sizeof(float));
    float* yS   = (float*)alloc((size_t)S_ * 2 * I_ * sizeof(float));

    quant_weight(w_qkv_s,         DDb, qQS, wsc + 0,  part, stream);
    quant_weight(w_qkv_c,         DDb, qQC, wsc + 2,  part, stream);
    quant_weight(w_ifg + 0 * DD_, DDb, qI,  wsc + 4,  part, stream);
    quant_weight(w_ifg + 1 * DD_, DDb, qF,  wsc + 6,  part, stream);
    quant_weight(w_ifg + 2 * DD_, DDb, qG,  wsc + 8,  part, stream);
    quant_weight(w_o,             DDb, qO,  wsc + 10, part, stream);
    quant_weight(w_out_s,         DDb, qOS, wsc + 12, part, stream);
    quant_weight(w_out_c,         DDb, qOC, wsc + 14, part, stream);
    quant_weight(w_gate_s,        GDb, qGS, wsc + 16, part, stream);
    quant_weight(w_down_s,        DIb, qDS, wsc + 18, part, stream);
    quant_weight(w_gate_c,        GDb, qGC, wsc + 20, part, stream);
    quant_weight(w_down_c,        DIb, qDC, wsc + 22, part, stream);

    silu_vec_kernel<<<(D_ + 255) / 256, 256, 0, stream>>>(temb, st_, D_);
    embed_kernel<<<(6 * D_ + 255) / 256, 256, 0, stream>>>(st_, w_emb_s, b_emb_s, e_s);
    embed_kernel<<<(6 * D_ + 255) / 256, 256, 0, stream>>>(st_, w_emb_c, b_emb_c, e_c);

    ln_mod_kernel<<<S_, 256, 0, stream>>>(hidden, e_s + 0 * D_, e_s + 1 * D_, nh);
    ln_mod_kernel<<<C_, 256, 0, stream>>>(enc,    e_c + 0 * D_, e_c + 1 * D_, nc);

    rms_quant_kernel<<<S_, 256, 0, stream>>>(nh, D_, D_, aq, aSc, 1e-8f);
    gemm_i8(aq, aSc, qQS, wsc + 1, qj + (size_t)C_ * D_, nullptr, nullptr, S_, D_, D_, stream);
    rms_quant_kernel<<<C_, 256, 0, stream>>>(nc, D_, D_, aq, aSc, 1e-8f);
    gemm_i8(aq, aSc, qQC, wsc + 3, qj, nullptr, nullptr, C_, D_, D_, stream);

    rms_quant_kernel<<<T_, 256, 0, stream>>>(qj, D_, D_, aq, aSc, 1e-8f);
    gemm_i8(aq, aSc, qI, wsc + 5, iL, nullptr, nullptr, T_, D_, D_, stream);
    gemm_i8(aq, aSc, qF, wsc + 7, fL, nullptr, nullptr, T_, D_, D_, stream);
    gemm_i8(aq, aSc, qG, wsc + 9, gL, nullptr, nullptr, T_, D_, D_, stream);

    hgrn_kernel<<<D_ / 256, 256, 0, stream>>>(iL, fL, oB);
    float* vB = iL;
    gated_kernel<<<T_, 256, 0, stream>>>(oB, gL, gnw, vB);

    float* attn = fL;
    rms_quant_kernel<<<T_, 256, 0, stream>>>(vB, D_, D_, aq, aSc, 1e-8f);
    gemm_i8(aq, aSc, qO, wsc + 11, attn, nullptr, nullptr, T_, D_, D_, stream);

    rms_quant_kernel<<<S_, 256, 0, stream>>>(attn + (size_t)C_ * D_, D_, D_, aq, aSc, 1e-8f);
    gemm_i8(aq, aSc, qOS, wsc + 13, out, hidden, e_s + 2 * D_, S_, D_, D_, stream);
    rms_quant_kernel<<<C_, 256, 0, stream>>>(attn, D_, D_, aq, aSc, 1e-8f);
    gemm_i8(aq, aSc, qOC, wsc + 15, out + (size_t)S_ * D_, enc, e_c + 2 * D_, C_, D_, D_, stream);

    ln_mod_kernel<<<S_, 256, 0, stream>>>(out, e_s + 3 * D_, e_s + 4 * D_, nh);
    rms_quant_kernel<<<S_, 256, 0, stream>>>(nh, D_, D_, aq, aSc, 1e-8f);
    gemm_i8(aq, aSc, qGS, wsc + 17, yS, nullptr, nullptr, S_, 2 * I_, D_, stream);
    glu_kernel<<<4096, 256, 0, stream>>>(yS, S_);
    rms_quant_kernel<<<S_, 256, 0, stream>>>(yS, 2 * I_, I_, aq, aSc, 1e-8f);
    gemm_i8(aq, aSc, qDS, wsc + 19, out, out, e_s + 5 * D_, S_, D_, I_, stream);

    float* outC = out + (size_t)S_ * D_;
    ln_mod_kernel<<<C_, 256, 0, stream>>>(outC, e_c + 3 * D_, e_c + 4 * D_, nc);
    rms_quant_kernel<<<C_, 256, 0, stream>>>(nc, D_, D_, aq, aSc, 1e-8f);
    gemm_i8(aq, aSc, qGC, wsc + 21, yS, nullptr, nullptr, C_, 2 * I_, D_, stream);
    glu_kernel<<<1024, 256, 0, stream>>>(yS, C_);
    rms_quant_kernel<<<C_, 256, 0, stream>>>(yS, 2 * I_, I_, aq, aSc, 1e-8f);
    gemm_i8(aq, aSc, qDC, wsc + 23, outC, outC, e_c + 5 * D_, C_, D_, I_, stream);
}
